// SimVQ_85796266705419
// MI455X (gfx1250) — matrix operand order repaired and checked by reading, not compiled, not run on hardware
//
#include <hip/hip_runtime.h>
#include <hip/hip_bf16.h>

typedef __attribute__((ext_vector_type(16))) __bf16 v16bf;
typedef __attribute__((ext_vector_type(8)))  __bf16 v8bf;
typedef __attribute__((ext_vector_type(4)))  __bf16 v4bf;
typedef __attribute__((ext_vector_type(8)))  float  v8f;

#define DDIM    512
#define M_TOK   8192
#define K_CODES 16384
#define KCH     16      // DDIM / 32 K-chunks per WMMA sweep
#define ROWS_PB 32      // z rows per dist block (2 row-tiles)

static __device__ __forceinline__ v16bf cat8(v8bf lo, v8bf hi) {
    return __builtin_shufflevector(lo, hi, 0,1,2,3,4,5,6,7,8,9,10,11,12,13,14,15);
}
static __device__ __forceinline__ v16bf ldb16(const __bf16* p) {
    return cat8(*(const v8bf*)p, *(const v8bf*)(p + 8));
}
static __device__ __forceinline__ void wmma_bf16(v16bf a, v16bf b, v8f& acc) {
    acc = __builtin_amdgcn_wmma_f32_16x16x32_bf16(false, a, false, b,
                                                  (short)0, acc, false, false);
}

// ---- convert proj_w to bf16, zero the loss accumulator -------------------
__global__ void prep_kernel(const float* __restrict__ w, __bf16* __restrict__ wbf,
                            float* __restrict__ accum) {
    int i = blockIdx.x * blockDim.x + threadIdx.x;
    if (i < DDIM * DDIM) wbf[i] = (__bf16)w[i];
    if (i == 0) accum[0] = 0.0f;
}

// ---- L2-normalize rows of src (f32 [rows,512]) into dst (bf16) -----------
__global__ void normalize_rows_kernel(const float* __restrict__ src,
                                      __bf16* __restrict__ dst, int rows) {
    int wave = threadIdx.x >> 5;
    int lane = threadIdx.x & 31;
    int row  = blockIdx.x * 4 + wave;
    if (row >= rows) return;
    const float4* s = (const float4*)(src + (size_t)row * DDIM);
    float4 v[4];
    float ss = 0.0f;
    #pragma unroll
    for (int j = 0; j < 4; ++j) {
        v[j] = s[lane + 32 * j];
        ss += v[j].x*v[j].x + v[j].y*v[j].y + v[j].z*v[j].z + v[j].w*v[j].w;
    }
    #pragma unroll
    for (int m = 16; m >= 1; m >>= 1) ss += __shfl_xor(ss, m, 32);
    float inv = 1.0f / fmaxf(sqrtf(ss), 1e-12f);
    __bf16* drow = dst + (size_t)row * DDIM;
    #pragma unroll
    for (int j = 0; j < 4; ++j) {
        v4bf o;
        o[0] = (__bf16)(v[j].x * inv); o[1] = (__bf16)(v[j].y * inv);
        o[2] = (__bf16)(v[j].z * inv); o[3] = (__bf16)(v[j].w * inv);
        *(v4bf*)(drow + (size_t)(lane + 32 * j) * 4) = o;
    }
}

// ---- qcb[16384,512] = E @ W^T + b via v_wmma_f32_16x16x32_bf16 -----------
// Block = one 16-row M-tile staged bf16 in LDS (shared by 4 waves); each
// wave sweeps 8 of the 32 N-tiles with double-buffered B fragments.
__global__ __launch_bounds__(128, 1)
void proj_gemm_kernel(const float* __restrict__ emb,
                      const __bf16* __restrict__ wbf,
                      const float* __restrict__ bias,
                      float* __restrict__ qcb) {
    __shared__ __bf16 et[16 * DDIM];          // 16 KB bf16 E tile
    int t     = threadIdx.x;
    int wave  = t >> 5;
    int lane  = t & 31;
    int mbase = blockIdx.x * 16;

    // cooperative stage: 16x512 f32 -> bf16 LDS
    {
        const float4* src = (const float4*)(emb + (size_t)mbase * DDIM);
        v4bf* dst = (v4bf*)et;
        #pragma unroll
        for (int j = 0; j < 16; ++j) {
            float4 f = src[t + 128 * j];
            v4bf o;
            o[0] = (__bf16)f.x; o[1] = (__bf16)f.y;
            o[2] = (__bf16)f.z; o[3] = (__bf16)f.w;
            dst[t + 128 * j] = o;
        }
    }
    __syncthreads();

    int klo0 = (lane & 16) ? 16 : 0;
    const __bf16* arow = et + (size_t)(lane & 15) * DDIM + klo0;
    int bkoff0 = (lane >> 4) * 16;

    for (int nt = wave * 8; nt < wave * 8 + 8; ++nt) {
        int col = nt * 16 + (lane & 15);
        const __bf16* wrow = wbf + (size_t)col * DDIM + bkoff0;
        v8f acc = {};
        v16bf b = ldb16(wrow);                      // prime kc = 0
        #pragma unroll
        for (int kc = 0; kc < KCH; ++kc) {
            v16bf a = cat8(*(const v8bf*)(arow + kc * 32),
                           *(const v8bf*)(arow + kc * 32 + 8));
            v16bf nb = b;
            if (kc + 1 < KCH) nb = ldb16(wrow + (kc + 1) * 32);  // prefetch
            wmma_bf16(a, b, acc);
            b = nb;
        }
        float bv    = bias[col];
        int   rbase = mbase + ((lane & 16) ? 8 : 0);
        #pragma unroll
        for (int r = 0; r < 8; ++r)
            qcb[(size_t)(rbase + r) * DDIM + col] = acc[r] + bv;
    }
}

// ---- fused distance GEMM + per-row argmax --------------------------------
// Block = 32 z-rows (two 16-row A tiles in LDS, shared by 4 waves). Each
// wave runs a 2x2 register tile: 2 row-tiles x 2 code-tiles = 4 accumulators,
// so every A and B fragment feeds two WMMAs. Scores never touch memory;
// c_n (16 MB) stays L2-resident and total L2 traffic is halved vs 16-row
// blocks (256 blocks x 16 MB).
__global__ __launch_bounds__(128, 1)
void dist_argmin_kernel(const __bf16* __restrict__ zn,
                        const __bf16* __restrict__ cn,
                        int* __restrict__ idx_out,
                        float* __restrict__ out_idx_f) {
    __shared__ __bf16 zt[ROWS_PB * DDIM];     // 32 KB bf16 z tile
    __shared__ float sv[4][ROWS_PB];
    __shared__ int   si[4][ROWS_PB];
    int t     = threadIdx.x;
    int wave  = t >> 5;
    int lane  = t & 31;
    int mbase = blockIdx.x * ROWS_PB;

    {   // cooperative stage: 32x512 bf16 = 2048 uint4
        const uint4* src = (const uint4*)(zn + (size_t)mbase * DDIM);
        uint4* dst = (uint4*)zt;
        #pragma unroll
        for (int j = 0; j < 16; ++j) dst[t + 128 * j] = src[t + 128 * j];
    }
    __syncthreads();

    float bestv[2][8];
    int   besti[2][8];
    #pragma unroll
    for (int rt = 0; rt < 2; ++rt)
        #pragma unroll
        for (int r = 0; r < 8; ++r) { bestv[rt][r] = -1e30f; besti[rt][r] = 0; }

    int klo0 = (lane & 16) ? 16 : 0;
    const __bf16* arow0 = zt + (size_t)(lane & 15) * DDIM + klo0;   // rows 0..15
    const __bf16* arow1 = arow0 + 16 * DDIM;                         // rows 16..31
    int bkoff0 = (lane >> 4) * 16;

    for (int p = wave; p < K_CODES / 32; p += 4) {
        int code0 = p * 32 + (lane & 15);
        const __bf16* c0 = cn + (size_t)code0 * DDIM + bkoff0;
        const __bf16* c1 = c0 + 16 * DDIM;
        v8f acc00 = {}, acc01 = {}, acc10 = {}, acc11 = {};
        v16bf b0 = ldb16(c0);                       // prime kc = 0
        v16bf b1 = ldb16(c1);
        #pragma unroll
        for (int kc = 0; kc < KCH; ++kc) {
            v16bf a0 = cat8(*(const v8bf*)(arow0 + kc * 32),
                            *(const v8bf*)(arow0 + kc * 32 + 8));
            v16bf a1 = cat8(*(const v8bf*)(arow1 + kc * 32),
                            *(const v8bf*)(arow1 + kc * 32 + 8));
            v16bf nb0 = b0, nb1 = b1;
            if (kc + 1 < KCH) {                     // prefetch next chunk
                nb0 = ldb16(c0 + (kc + 1) * 32);
                nb1 = ldb16(c1 + (kc + 1) * 32);
            }
            wmma_bf16(a0, b0, acc00);
            wmma_bf16(a0, b1, acc01);
            wmma_bf16(a1, b0, acc10);
            wmma_bf16(a1, b1, acc11);
            b0 = nb0; b1 = nb1;
        }
        #pragma unroll
        for (int r = 0; r < 8; ++r) {               // code grows per wave:
            float s;                                // '>' keeps first match
            s = acc00[r]; if (s > bestv[0][r]) { bestv[0][r] = s; besti[0][r] = code0;      }
            s = acc01[r]; if (s > bestv[0][r]) { bestv[0][r] = s; besti[0][r] = code0 + 16; }
            s = acc10[r]; if (s > bestv[1][r]) { bestv[1][r] = s; besti[1][r] = code0;      }
            s = acc11[r]; if (s > bestv[1][r]) { bestv[1][r] = s; besti[1][r] = code0 + 16; }
        }
    }

    // butterfly argmax across the 16 lanes sharing each row group
    #pragma unroll
    for (int m = 8; m >= 1; m >>= 1) {
        #pragma unroll
        for (int rt = 0; rt < 2; ++rt) {
            #pragma unroll
            for (int r = 0; r < 8; ++r) {
                float ov = __shfl_xor(bestv[rt][r], m, 32);
                int   oi = __shfl_xor(besti[rt][r], m, 32);
                if (ov > bestv[rt][r] ||
                    (ov == bestv[rt][r] && oi < besti[rt][r])) {
                    bestv[rt][r] = ov; besti[rt][r] = oi;
                }
            }
        }
    }
    if (lane == 0 || lane == 16) {
        int half = (lane >> 4) * 8;
        #pragma unroll
        for (int rt = 0; rt < 2; ++rt)
            #pragma unroll
            for (int r = 0; r < 8; ++r) {
                sv[wave][rt * 16 + half + r] = bestv[rt][r];
                si[wave][rt * 16 + half + r] = besti[rt][r];
            }
    }
    __syncthreads();
    if (t < ROWS_PB) {
        float bv = sv[0][t];
        int   bi = si[0][t];
        #pragma unroll
        for (int w = 1; w < 4; ++w) {
            float ov = sv[w][t];
            int   oi = si[w][t];
            if (ov > bv || (ov == bv && oi < bi)) { bv = ov; bi = oi; }
        }
        idx_out[mbase + t]   = bi;
        out_idx_f[mbase + t] = (float)bi;
    }
}

// ---- gather quantized rows + accumulate sum((q-z)^2) ---------------------
__global__ void gather_loss_kernel(const float* __restrict__ z,
                                   const float* __restrict__ qcb,
                                   const int* __restrict__ idx,
                                   float* __restrict__ out_q,
                                   float* __restrict__ accum) {
    __shared__ float red[4];
    int row = blockIdx.x;
    int t   = threadIdx.x;
    int e   = idx[row];
    const float4* q  = (const float4*)(qcb + (size_t)e   * DDIM);
    const float4* zz = (const float4*)(z   + (size_t)row * DDIM);
    float4*       o  = (float4*)(out_q + (size_t)row * DDIM);
    float4 qv = q[t];
    float4 zv = zz[t];
    o[t] = qv;
    float dx = qv.x - zv.x, dy = qv.y - zv.y, dz = qv.z - zv.z, dw = qv.w - zv.w;
    float s = dx*dx + dy*dy + dz*dz + dw*dw;
    #pragma unroll
    for (int m = 16; m >= 1; m >>= 1) s += __shfl_xor(s, m, 32);
    if ((t & 31) == 0) red[t >> 5] = s;
    __syncthreads();
    if (t == 0) atomicAdd(accum, red[0] + red[1] + red[2] + red[3]);
}

__global__ void finalize_kernel(const float* __restrict__ accum,
                                float* __restrict__ out_loss) {
    out_loss[0] = accum[0] * 1.25f / (float)((size_t)M_TOK * DDIM);
}

extern "C" void kernel_launch(void* const* d_in, const int* in_sizes, int n_in,
                              void* d_out, int out_size, void* d_ws, size_t ws_size,
                              hipStream_t stream) {
    (void)in_sizes; (void)n_in; (void)out_size; (void)ws_size;
    const float* z   = (const float*)d_in[0];   // [8,1024,512]
    const float* emb = (const float*)d_in[1];   // [16384,512]
    const float* pw  = (const float*)d_in[2];   // [512,512]
    const float* pb  = (const float*)d_in[3];   // [512]
    // d_in[4] = scale: positive scalar, argmin-invariant -> unused

    float* out      = (float*)d_out;
    float* out_q    = out;                          // 8192*512
    float* out_loss = out + (size_t)M_TOK * DDIM;   // 1
    float* out_idx  = out_loss + 1;                 // 8192 (as float)

    char*   ws    = (char*)d_ws;                    // ~59.3 MB used
    float*  qcb   = (float*)ws;                     // 32 MB f32 codebook
    __bf16* cn    = (__bf16*)(ws + 33554432);       // 16 MB normalized codes
    __bf16* zn    = (__bf16*)(ws + 50331648);       //  8 MB normalized z
    __bf16* wbf   = (__bf16*)(ws + 58720256);       // 0.5 MB W bf16
    int*    idx   = (int*)  (ws + 59244544);        // 32 KB indices
    float*  accum = (float*)(ws + 59277312);        // loss accumulator

    prep_kernel<<<(DDIM * DDIM) / 256, 256, 0, stream>>>(pw, wbf, accum);
    normalize_rows_kernel<<<M_TOK / 4,   128, 0, stream>>>(z,   zn, M_TOK);
    proj_gemm_kernel<<<K_CODES / 16,     128, 0, stream>>>(emb, wbf, pb, qcb);
    normalize_rows_kernel<<<K_CODES / 4, 128, 0, stream>>>(qcb, cn, K_CODES);
    dist_argmin_kernel<<<M_TOK / ROWS_PB, 128, 0, stream>>>(zn, cn, idx, out_idx);
    gather_loss_kernel<<<M_TOK,          128, 0, stream>>>(z, qcb, idx, out_q, accum);
    finalize_kernel<<<1, 1, 0, stream>>>(accum, out_loss);
}
